// PathOptimizerGCN_45346264711272
// MI455X (gfx1250) — matrix operand order repaired and checked by reading, not compiled, not run on hardware
//
#include <hip/hip_runtime.h>
#include <hip/hip_bf16.h>
#include <math.h>

// ---------------------------------------------------------------------------
// Types for CDNA5 WMMA (wave32, bf16 A/B, f32 accumulate)
// ---------------------------------------------------------------------------
typedef __bf16 bf16_t;
typedef bf16_t v16bf __attribute__((ext_vector_type(16)));
typedef bf16_t v8bf  __attribute__((ext_vector_type(8)));
typedef bf16_t v4bf  __attribute__((ext_vector_type(4)));
typedef float  v8f   __attribute__((ext_vector_type(8)));

#define FEAT   64      // node feature dim (D == H == 64)
#define EDIM   16      // edge_attr dim
#define KEDGE  160     // 144 padded to a multiple of 32

// ---------------------------------------------------------------------------
// WMMA fragment loaders (contiguous k = 16h + s per lane half, wave32)
// ---------------------------------------------------------------------------
__device__ __forceinline__ v16bf load_a_frag(const bf16_t* rowPtr, int kb, int lane) {
  const int off = (lane < 16) ? 0 : 16;
  v8bf lo = *(const v8bf*)(rowPtr + kb + off);        // ds_load_b128
  v8bf hi = *(const v8bf*)(rowPtr + kb + off + 8);    // ds_load_b128
  v16bf a;
#pragma unroll
  for (int j = 0; j < 8; ++j) { a[j] = lo[j]; a[j + 8] = hi[j]; }
  return a;
}

__device__ __forceinline__ v16bf load_b_frag(const bf16_t* colPtr, int kb, int lane) {
  const int off = (lane < 16) ? 0 : 16;
  return *(const v16bf*)(colPtr + kb + off);          // 2x ds_load_b128
}

__device__ __forceinline__ v4bf cvt4(float4 v) {
  v4bf r;
  r[0] = (bf16_t)v.x; r[1] = (bf16_t)v.y; r[2] = (bf16_t)v.z; r[3] = (bf16_t)v.w;
  return r;
}

// ---------------------------------------------------------------------------
// One-time weight prep: bf16, transposed (K-contiguous per output column).
//   w1t : [64][64]   W1t[n][k]  = W1[k*64+n]
//   wt  : [64][160]  Wt[n][k]   = Wm1[k*64+n], zero-padded k in [144,160)
// ---------------------------------------------------------------------------
__global__ __launch_bounds__(256) void k_prep_w(const float* __restrict__ W1,
                                                const float* __restrict__ Wm1,
                                                bf16_t* __restrict__ w1t,
                                                bf16_t* __restrict__ wt) {
  int i = blockIdx.x * blockDim.x + threadIdx.x;
  if (i < FEAT * KEDGE) {
    int n = i / KEDGE, k = i - n * KEDGE;
    wt[i] = (bf16_t)((k < 2 * FEAT + EDIM) ? Wm1[k * FEAT + n] : 0.0f);
  }
  int j = i - FEAT * KEDGE;
  if (j >= 0 && j < FEAT * FEAT) {
    int n = j >> 6, k = j & 63;
    w1t[j] = (bf16_t)W1[k * FEAT + n];
  }
}

// ---------------------------------------------------------------------------
// Degree / normalization kernels
// ---------------------------------------------------------------------------
__global__ __launch_bounds__(256) void k_deg_init(float* deg, int N) {
  int i = blockIdx.x * blockDim.x + threadIdx.x;
  if (i < N) deg[i] = 1.0f;   // self loop contributes 1
}

__global__ __launch_bounds__(256) void k_deg_count(const int* dst, float* deg, int E) {
  int i = blockIdx.x * blockDim.x + threadIdx.x;
  if (i < E) atomicAdd(&deg[dst[i]], 1.0f);
}

__global__ __launch_bounds__(256) void k_dinv(const float* deg, float* dinv, int N) {
  int i = blockIdx.x * blockDim.x + threadIdx.x;
  if (i < N) {
    float d = deg[i];
    dinv[i] = (d > 0.0f) ? rsqrtf(d) : 0.0f;
  }
}

// ---------------------------------------------------------------------------
// h0 = x @ W1  (N x 64 @ 64 x 64), bf16 WMMA, one 16x64 tile per wave.
// C fragment is transposed through LDS so global stores are b128-coalesced.
// ---------------------------------------------------------------------------
__global__ __launch_bounds__(256) void k_gemm_xw(const float* __restrict__ x,
                                                 const bf16_t* __restrict__ w1t,
                                                 float* __restrict__ h0, int N) {
  __shared__ __attribute__((aligned(32))) bf16_t W1t[FEAT][FEAT];   // 8 KB
  __shared__ __attribute__((aligned(32))) bf16_t Xt[8][16][FEAT];   // 16 KB
  __shared__ __attribute__((aligned(16))) float  Ot[8][16][FEAT];   // 32 KB

  const int tid  = threadIdx.x;
  const int lane = tid & 31;
  const int wv   = tid >> 5;

  // stage pre-transposed W1 (contiguous b128 copies)
  {
    const uint4* srcp = (const uint4*)w1t;
    uint4* dstp = (uint4*)&W1t[0][0];
    for (int i = tid; i < (FEAT * FEAT) / 8; i += 256) dstp[i] = srcp[i];
  }

  const int rowBase = (blockIdx.x * 8 + wv) * 16;
  const bool tileValid = (rowBase < N);
  const bool fullTile  = (rowBase + 16 <= N);

  // stage this wave's 16x64 A tile as bf16 (float4 loads, b64 LDS stores)
  for (int i = lane; i < 16 * (FEAT / 4); i += 32) {   // 16 rows x 16 float4
    int r = i >> 4, c4 = i & 15;
    int row = rowBase + r;
    float4 v = make_float4(0.f, 0.f, 0.f, 0.f);
    if (row < N) v = *(const float4*)(x + (size_t)row * FEAT + c4 * 4);
    *(v4bf*)&Xt[wv][r][c4 * 4] = cvt4(v);
  }
  __syncthreads();

  v8f acc[4] = {};
  if (tileValid) {
    const bf16_t* rowPtr = &Xt[wv][lane & 15][0];
#pragma unroll
    for (int kb = 0; kb < FEAT; kb += 32) {
      v16bf a = load_a_frag(rowPtr, kb, lane);
#pragma unroll
      for (int t = 0; t < 4; ++t) {
        v16bf b = load_b_frag(&W1t[t * 16 + (lane & 15)][0], kb, lane);
        acc[t] = __builtin_amdgcn_wmma_f32_16x16x32_bf16(
            false, a, false, b, (short)0, acc[t], false, false);
      }
    }
  }

  // transpose C through LDS (wave-private region, lockstep within wave)
  const int mOff = (lane < 16) ? 0 : 8;
  const int nCol = lane & 15;
#pragma unroll
  for (int t = 0; t < 4; ++t)
#pragma unroll
    for (int r = 0; r < 8; ++r)
      Ot[wv][r + mOff][t * 16 + nCol] = acc[t][r];

  if (tileValid) {
    for (int i = lane; i < 16 * (FEAT / 4); i += 32) {  // 16 rows x 16 float4
      int r = i >> 4, c4 = i & 15;
      int row = rowBase + r;
      if (fullTile || row < N)
        *(float4*)(h0 + (size_t)row * FEAT + c4 * 4) = *(const float4*)&Ot[wv][r][c4 * 4];
    }
  }
}

// ---------------------------------------------------------------------------
// agg init with self loops: agg[i] = h0[i] * dinv[i]^2
// ---------------------------------------------------------------------------
__global__ __launch_bounds__(256) void k_self(const float* __restrict__ h0,
                                              const float* __restrict__ dinv,
                                              float* __restrict__ agg, int N) {
  int i = blockIdx.x * blockDim.x + threadIdx.x;
  if (i < N * FEAT) {
    int node = i >> 6;
    float dv = dinv[node];
    agg[i] = h0[i] * dv * dv;
  }
}

// ---------------------------------------------------------------------------
// edge scatter: agg[dst] += h0[src] * dinv[src]*dinv[dst]
// ---------------------------------------------------------------------------
__global__ __launch_bounds__(256) void k_scatter(const float* __restrict__ h0,
                                                 const float* __restrict__ dinv,
                                                 const int* __restrict__ src,
                                                 const int* __restrict__ dst,
                                                 float* __restrict__ agg, int E) {
  long long i = (long long)blockIdx.x * blockDim.x + threadIdx.x;
  if (i < (long long)E * FEAT) {
    int e = (int)(i >> 6), f = (int)(i & 63);
    int s = src[e], d = dst[e];
    float nrm = dinv[s] * dinv[d];
    atomicAdd(&agg[(size_t)d * FEAT + f], h0[(size_t)s * FEAT + f] * nrm);
  }
}

// ---------------------------------------------------------------------------
// h1 = relu(agg + b1) (in place)
// ---------------------------------------------------------------------------
__global__ __launch_bounds__(256) void k_bias_relu(float* __restrict__ agg,
                                                   const float* __restrict__ b1, int N) {
  int i = blockIdx.x * blockDim.x + threadIdx.x;
  if (i < N * FEAT) agg[i] = fmaxf(agg[i] + b1[i & 63], 0.0f);
}

// ---------------------------------------------------------------------------
// Edge MLP: scores = sigmoid(relu([h1[src]|h1[dst]|ea] @ Wm1 + bm1) @ Wm2 + bm2)
// 16 edges per wave; K padded 144 -> 160; 20 bf16 WMMAs per wave; layer 2
// fused in-register (shfl_xor butterfly + register select + coalesced store).
// ---------------------------------------------------------------------------
__global__ __launch_bounds__(256) void k_edge_mlp(const float* __restrict__ h1,
                                                  const float* __restrict__ ea,
                                                  const int* __restrict__ src,
                                                  const int* __restrict__ dst,
                                                  const bf16_t* __restrict__ wt,
                                                  const float* __restrict__ bm1,
                                                  const float* __restrict__ Wm2,
                                                  const float* __restrict__ bm2,
                                                  float* __restrict__ out, int E) {
  __shared__ __attribute__((aligned(32))) bf16_t Wt[FEAT][KEDGE];      // 20 KB
  __shared__ __attribute__((aligned(32))) bf16_t At[8][16][KEDGE];     // 40 KB
  __shared__ float bm1s[FEAT];
  __shared__ float w2s[FEAT];

  const int tid  = threadIdx.x;
  const int lane = tid & 31;
  const int wv   = tid >> 5;

  // stage pre-transposed Wm1 (contiguous b128 copies: 5 per thread)
  {
    const uint4* srcp = (const uint4*)wt;
    uint4* dstp = (uint4*)&Wt[0][0];
    for (int i = tid; i < (FEAT * KEDGE) / 8; i += 256) dstp[i] = srcp[i];
  }
  if (tid < FEAT) { bm1s[tid] = bm1[tid]; w2s[tid] = Wm2[tid]; }

  const int edgeBase = (blockIdx.x * 8 + wv) * 16;
  // stage this wave's 16x160 bf16 tile: float4 gathers, b64 LDS stores
  for (int i = lane; i < 16 * (KEDGE / 4); i += 32) {   // 16 rows x 40 float4
    int r = i / 40, c4 = i - r * 40;
    int e = edgeBase + r;
    float4 v = make_float4(0.f, 0.f, 0.f, 0.f);
    if (e < E) {
      if (c4 < 16)      v = *(const float4*)(h1 + (size_t)src[e] * FEAT + c4 * 4);
      else if (c4 < 32) v = *(const float4*)(h1 + (size_t)dst[e] * FEAT + (c4 - 16) * 4);
      else if (c4 < 36) v = *(const float4*)(ea + (size_t)e * EDIM + (c4 - 32) * 4);
    }
    *(v4bf*)&At[wv][r][c4 * 4] = cvt4(v);
  }
  __syncthreads();

  v8f acc[4] = {};
  const bf16_t* rowPtr = &At[wv][lane & 15][0];
#pragma unroll
  for (int kb = 0; kb < KEDGE; kb += 32) {
    v16bf a = load_a_frag(rowPtr, kb, lane);
#pragma unroll
    for (int t = 0; t < 4; ++t) {
      v16bf b = load_b_frag(&Wt[t * 16 + (lane & 15)][0], kb, lane);
      acc[t] = __builtin_amdgcn_wmma_f32_16x16x32_bf16(
          false, a, false, b, (short)0, acc[t], false, false);
    }
  }

  // layer 2: relu(hidden + bm1) . Wm2, reduced across the 16 N-columns
  const float bm2v = bm2[0];
  float partial[8] = {0.f, 0.f, 0.f, 0.f, 0.f, 0.f, 0.f, 0.f};
#pragma unroll
  for (int t = 0; t < 4; ++t) {
    int n = t * 16 + (lane & 15);
    float bias = bm1s[n];
    float w2   = w2s[n];
#pragma unroll
    for (int r = 0; r < 8; ++r) {
      float hv = fmaxf(acc[t][r] + bias, 0.0f);
      partial[r] += hv * w2;
    }
  }
#pragma unroll
  for (int r = 0; r < 8; ++r) {
    partial[r] += __shfl_xor(partial[r], 1, 32);
    partial[r] += __shfl_xor(partial[r], 2, 32);
    partial[r] += __shfl_xor(partial[r], 4, 32);
    partial[r] += __shfl_xor(partial[r], 8, 32);
  }
  // lanes 0-15 hold rows 0-7 (all equal per half); lanes 16-31 hold rows 8-15.
  // Select own element, route rows 8-15 down from the upper half, store
  // one coalesced b32 from lanes 0-15.
  float sel = partial[0];
#pragma unroll
  for (int r = 1; r < 8; ++r) sel = ((lane & 7) == r) ? partial[r] : sel;
  int srcLane = (lane < 8) ? lane : lane + 8;
  float v = __shfl(sel, srcLane, 32);
  if (lane < 16) {
    int e = edgeBase + lane;
    if (e < E) out[e] = 1.0f / (1.0f + __expf(-(v + bm2v)));
  }
}

// ---------------------------------------------------------------------------
// Launch
// ---------------------------------------------------------------------------
extern "C" void kernel_launch(void* const* d_in, const int* in_sizes, int n_in,
                              void* d_out, int out_size, void* d_ws, size_t ws_size,
                              hipStream_t stream) {
  const float* x   = (const float*)d_in[0];
  const int*   src = (const int*)d_in[1];
  const int*   dst = (const int*)d_in[2];
  const float* ea  = (const float*)d_in[3];
  const float* W1  = (const float*)d_in[4];
  const float* b1  = (const float*)d_in[5];
  const float* Wm1 = (const float*)d_in[6];
  const float* bm1 = (const float*)d_in[7];
  const float* Wm2 = (const float*)d_in[8];
  const float* bm2 = (const float*)d_in[9];
  float* out = (float*)d_out;

  const int N = in_sizes[0] / FEAT;   // 50000
  const int E = in_sizes[1];          // 800000

  float* ws   = (float*)d_ws;
  float* deg  = ws;
  float* dinv = deg + N;
  float* h0   = dinv + N;
  float* agg  = h0 + (size_t)N * FEAT;
  bf16_t* w1t = (bf16_t*)(agg + (size_t)N * FEAT);            // 64*64 bf16
  bf16_t* wt  = w1t + FEAT * FEAT;                            // 64*160 bf16

  const int T = 256;
  // one-time weight transpose/convert
  k_prep_w   <<<(FEAT * KEDGE + FEAT * FEAT + T - 1) / T, T, 0, stream>>>(W1, Wm1, w1t, wt);

  // degree + normalization
  k_deg_init <<<(N + T - 1) / T, T, 0, stream>>>(deg, N);
  k_deg_count<<<(E + T - 1) / T, T, 0, stream>>>(dst, deg, E);
  k_dinv     <<<(N + T - 1) / T, T, 0, stream>>>(deg, dinv, N);

  // h0 = x @ W1 (WMMA)
  int rowTiles = (N + 15) / 16;
  k_gemm_xw  <<<(rowTiles + 7) / 8, T, 0, stream>>>(x, w1t, h0, N);

  // normalized aggregation (self loops + edges), bias + relu
  k_self     <<<((N * FEAT) + T - 1) / T, T, 0, stream>>>(h0, dinv, agg, N);
  long long ew = (long long)E * FEAT;
  k_scatter  <<<(int)((ew + T - 1) / T), T, 0, stream>>>(h0, dinv, src, dst, agg, E);
  k_bias_relu<<<((N * FEAT) + T - 1) / T, T, 0, stream>>>(agg, b1, N);

  // fused edge MLP (WMMA + in-register layer 2)
  int edgeTiles = (E + 15) / 16;
  k_edge_mlp <<<(edgeTiles + 7) / 8, T, 0, stream>>>(agg, ea, src, dst,
                                                     wt, bm1, Wm2, bm2, out, E);
}
